// MixtureOfExperts_78958678769762
// MI455X (gfx1250) — matrix operand order repaired and checked by reading, not compiled, not run on hardware
//
#include <hip/hip_runtime.h>
#include <math.h>

// ---------- types ----------
typedef __bf16 bf16_t;
typedef __attribute__((ext_vector_type(8)))  __bf16 v8bf;
typedef __attribute__((ext_vector_type(16))) __bf16 v16bf;
typedef __attribute__((ext_vector_type(8)))  float  v8f;

#define D_DIM 256
#define H_DIM 1024
#define E_NUM 8
#define MT_BLK 4                 // m-tiles per block (block covers 64 rows)
#define ROWS_BLK (16 * MT_BLK)

// meta layout (ints): [0..7] counts, [8..15] cursors, [16..24] slotBase,
// [32..40] blockBase, [48] totalBlocks, [49] totalPadSlots, [64..71] usage (as float)

__device__ __forceinline__ bf16_t f2bf(float f) {
    unsigned u = __builtin_bit_cast(unsigned, f);
    unsigned r = u + 0x7FFFu + ((u >> 16) & 1u);     // round-to-nearest-even
    unsigned short h = (unsigned short)(r >> 16);
    return __builtin_bit_cast(bf16_t, h);
}

__device__ __forceinline__ float gelu_exact(float v) {
    return 0.5f * v * (1.0f + erff(v * 0.70710678118654752f));
}

// ---------- init / fill ----------
__global__ void moe_init_meta(int* meta) {
    int i = threadIdx.x;
    if (i < 96) meta[i] = 0;
}

__global__ void moe_fill(int* tokList, bf16_t* zeroRow, int cap) {
    int i = blockIdx.x * blockDim.x + threadIdx.x;
    if (i < cap) tokList[i] = -1;
    if (i < H_DIM) zeroRow[i] = __builtin_bit_cast(bf16_t, (unsigned short)0);
}

__global__ void moe_f32_to_bf16(const float* __restrict__ in, bf16_t* __restrict__ out, size_t n) {
    size_t i = (size_t)blockIdx.x * blockDim.x + threadIdx.x;
    size_t stride = (size_t)gridDim.x * blockDim.x;
    for (; i < n; i += stride) out[i] = f2bf(in[i]);
}

// ---------- router: one wave32 per token ----------
__global__ void moe_router(const float* __restrict__ x, const float* __restrict__ gw,
                           float* __restrict__ gatesPair, int* __restrict__ expertPair,
                           int* __restrict__ meta, int N) {
    int wid  = (blockIdx.x * blockDim.x + threadIdx.x) >> 5;
    int lane = threadIdx.x & 31;
    if (wid >= N) return;
    const float* xr = x + (size_t)wid * D_DIM;
    float acc[E_NUM];
#pragma unroll
    for (int e = 0; e < E_NUM; ++e) acc[e] = 0.0f;
    for (int j = lane; j < D_DIM; j += 32) {
        float xv = xr[j];
#pragma unroll
        for (int e = 0; e < E_NUM; ++e) acc[e] += xv * gw[e * D_DIM + j];
    }
#pragma unroll
    for (int e = 0; e < E_NUM; ++e) {
#pragma unroll
        for (int off = 16; off > 0; off >>= 1) acc[e] += __shfl_xor(acc[e], off, 32);
    }
    if (lane == 0) {
        int i0 = 0; float v0 = acc[0];
#pragma unroll
        for (int e = 1; e < E_NUM; ++e) if (acc[e] > v0) { v0 = acc[e]; i0 = e; }
        int i1 = -1; float v1 = -3.402823e38f;
#pragma unroll
        for (int e = 0; e < E_NUM; ++e) if (e != i0 && acc[e] > v1) { v1 = acc[e]; i1 = e; }
        float g0 = 1.0f / (1.0f + expf(v1 - v0));   // softmax over {v0, v1}, v0 >= v1
        float g1 = 1.0f - g0;
        gatesPair[2 * wid]     = g0;
        gatesPair[2 * wid + 1] = g1;
        expertPair[2 * wid]     = i0;
        expertPair[2 * wid + 1] = i1;
        atomicAdd(&meta[i0], 1);
        atomicAdd(&meta[i1], 1);
    }
}

// ---------- exclusive scan over 8 experts (single thread, trivial) ----------
__global__ void moe_scan(int* meta) {
    if (blockIdx.x == 0 && threadIdx.x == 0) {
        int slot = 0, blk = 0;
        for (int e = 0; e < E_NUM; ++e) {
            meta[16 + e] = slot;
            meta[32 + e] = blk;
            int pn = (meta[e] + (ROWS_BLK - 1)) & ~(ROWS_BLK - 1);
            slot += pn;
            blk += pn / ROWS_BLK;
        }
        meta[16 + E_NUM] = slot;
        meta[32 + E_NUM] = blk;
        meta[48] = blk;
        meta[49] = slot;
    }
}

// ---------- scatter (token,choice) pairs into per-expert compacted slots ----------
__global__ void moe_assign(const int* __restrict__ expertPair, int* __restrict__ meta,
                           int* __restrict__ tokList, int* __restrict__ tokSlots, int n2) {
    int i = blockIdx.x * blockDim.x + threadIdx.x;
    if (i >= n2) return;
    int e = expertPair[i];
    int pos = atomicAdd(&meta[8 + e], 1);
    int slot = meta[16 + e] + pos;
    tokList[slot] = i >> 1;
    tokSlots[i] = slot;
}

// ---------- deterministic per-expert usage reduction ----------
__global__ void moe_usage(const float* __restrict__ gatesPair, const int* __restrict__ expertPair,
                          float* __restrict__ usage, int n2, int N) {
    __shared__ float sm[256];
    int e = blockIdx.x;
    float s = 0.0f;
    for (int i = threadIdx.x; i < n2; i += 256)
        if (expertPair[i] == e) s += gatesPair[i];
    sm[threadIdx.x] = s;
    __syncthreads();
    for (int k = 128; k > 0; k >>= 1) {
        if (threadIdx.x < k) sm[threadIdx.x] += sm[threadIdx.x + k];
        __syncthreads();
    }
    if (threadIdx.x == 0) usage[e] = sm[0] / (float)N;
}

__global__ void moe_loss(const float* __restrict__ usage, float* __restrict__ outLoss) {
    if (blockIdx.x == 0 && threadIdx.x == 0) {
        const float u = 1.0f / (float)E_NUM;
        const float logu = logf(u);
        float lb = 0.0f;
        for (int e = 0; e < E_NUM; ++e) {
            float lu = fmaxf(logf(usage[e]), -1e9f);
            lb += u * (logu - lu);
        }
        *outLoss = lb;
    }
}

// ---------- grouped GEMM with WMMA bf16, register-blocked 4Mx(NTC)N per wave ----------
// Block = 256 threads = 8 waves; block covers ROWS_BLK (=64) rows of one expert.
// Each wave owns NTC n-tiles per chunk across all MT_BLK m-tiles:
//   per K-step: MT_BLK A-frag loads + NTC B-frag loads feed MT_BLK*NTC WMMAs.
template <int KD, int ND, int NTC, bool GELU_OUT, bool GATHER, bool OUT_BF16>
__global__ __launch_bounds__(256) void moe_gemm(
        const bf16_t* __restrict__ Asrc, const bf16_t* __restrict__ W,
        const float* __restrict__ bias, void* __restrict__ Out,
        const int* __restrict__ tokList, const int* __restrict__ meta,
        const bf16_t* __restrict__ zeroRow) {
    int blk = blockIdx.x;
    if (blk >= meta[48]) return;             // beyond used blocks this launch
    int e = 0;
#pragma unroll
    for (int k = 1; k < E_NUM; ++k) if (blk >= meta[32 + k]) e = k;
    int mBase = meta[16 + e] + (blk - meta[32 + e]) * ROWS_BLK;

    int lane    = threadIdx.x & 31;
    int wave    = threadIdx.x >> 5;
    int halfsel = lane >> 4;                 // 0: lanes 0-15, 1: lanes 16-31
    int lrow    = lane & 15;

    // A-matrix row pointers, one per m-tile (row M = lrow within each 16-row tile)
    const bf16_t* aRow[MT_BLK];
#pragma unroll
    for (int mt = 0; mt < MT_BLK; ++mt) {
        int r = mBase + mt * 16 + lrow;
        if (GATHER) {
            int tok = tokList[r];
            aRow[mt] = (tok < 0) ? zeroRow : (Asrc + (size_t)tok * KD);
        } else {
            aRow[mt] = Asrc + (size_t)r * KD;
        }
    }
    const bf16_t* We = W + (size_t)e * ND * KD;
    const int kh8  = halfsel * 8;            // unused: A also uses kh16
    const int kh16 = halfsel * 16;           // B: K sub-offset per lane half

    constexpr int NCHUNK = ND / (16 * 8 * NTC);
#pragma unroll
    for (int c = 0; c < NCHUNK; ++c) {
        v8f acc[MT_BLK][NTC];
#pragma unroll
        for (int mt = 0; mt < MT_BLK; ++mt)
#pragma unroll
            for (int j = 0; j < NTC; ++j)
#pragma unroll
                for (int r = 0; r < 8; ++r) acc[mt][j][r] = 0.0f;

        int cols[NTC];
        const bf16_t* wR[NTC];
#pragma unroll
        for (int j = 0; j < NTC; ++j) {
            int tile = (c * 8 + wave) * NTC + j;
            cols[j] = tile * 16 + lrow;
            wR[j] = We + (size_t)cols[j] * KD;
        }

        for (int k0 = 0; k0 < KD; k0 += 32) {
            v16bf a[MT_BLK];
#pragma unroll
            for (int mt = 0; mt < MT_BLK; ++mt) {
                // A frag: elems 0-15 = K[k0+kh16..+15]
                v8bf lo = *(const v8bf*)(aRow[mt] + k0 + kh16);
                v8bf hi = *(const v8bf*)(aRow[mt] + k0 + kh16 + 8);
#pragma unroll
                for (int i = 0; i < 8; ++i) { a[mt][i] = lo[i]; a[mt][i + 8] = hi[i]; }
            }
#pragma unroll
            for (int j = 0; j < NTC; ++j) {
                // B frag: elems 0-15 = K[k0+kh16..+15] of column cols[j]
                v16bf b = *(const v16bf*)(wR[j] + k0 + kh16);
#pragma unroll
                for (int mt = 0; mt < MT_BLK; ++mt)
                    acc[mt][j] = __builtin_amdgcn_wmma_f32_16x16x32_bf16(
                        false, a[mt], false, b, (short)0, acc[mt][j], false, false);
            }
        }

        // epilogue: bias + activation + store
#pragma unroll
        for (int j = 0; j < NTC; ++j) {
            float bv = bias[e * ND + cols[j]];
#pragma unroll
            for (int mt = 0; mt < MT_BLK; ++mt) {
                size_t rowBase = (size_t)mBase + mt * 16 + halfsel * 8;
                if (OUT_BF16) {
                    bf16_t* o = (bf16_t*)Out;
#pragma unroll
                    for (int r = 0; r < 8; ++r) {
                        float v = acc[mt][j][r] + bv;
                        if (GELU_OUT) v = gelu_exact(v);
                        o[(rowBase + r) * ND + cols[j]] = f2bf(v);
                    }
                } else {
                    float* o = (float*)Out;
#pragma unroll
                    for (int r = 0; r < 8; ++r) {
                        float v = acc[mt][j][r] + bv;
                        if (GELU_OUT) v = gelu_exact(v);
                        o[(rowBase + r) * ND + cols[j]] = v;
                    }
                }
            }
        }
    }
}

// ---------- gated combine ----------
__global__ void moe_combine(const float* __restrict__ Y, const float* __restrict__ gatesPair,
                            const int* __restrict__ tokSlots, float* __restrict__ out, int total) {
    int idx = blockIdx.x * blockDim.x + threadIdx.x;
    if (idx >= total) return;
    int t   = idx >> 8;          // D_DIM == 256
    int col = idx & 255;
    int   s0 = tokSlots[2 * t],     s1 = tokSlots[2 * t + 1];
    float g0 = gatesPair[2 * t],    g1 = gatesPair[2 * t + 1];
    out[idx] = g0 * Y[(size_t)s0 * D_DIM + col] + g1 * Y[(size_t)s1 * D_DIM + col];
}

// ---------- host launcher ----------
extern "C" void kernel_launch(void* const* d_in, const int* in_sizes, int n_in,
                              void* d_out, int out_size, void* d_ws, size_t ws_size,
                              hipStream_t stream) {
    const float* x  = (const float*)d_in[0];
    const float* gw = (const float*)d_in[1];
    const float* w1 = (const float*)d_in[2];
    const float* b1 = (const float*)d_in[3];
    const float* w2 = (const float*)d_in[4];
    const float* b2 = (const float*)d_in[5];
    const float* w3 = (const float*)d_in[6];
    const float* b3 = (const float*)d_in[7];

    const int d = D_DIM, h = H_DIM, E = E_NUM;
    const int N  = in_sizes[0] / d;              // 16384 tokens
    const int N2 = 2 * N;                        // (token, choice) pairs
    const int padCap = N2 + ROWS_BLK * E;        // padded slot capacity

    // workspace carving (256B aligned)
    char* ws = (char*)d_ws;
    size_t off = 0;
    auto carve = [&](size_t bytes) -> char* {
        char* p = ws + off;
        off += (bytes + 255) & ~(size_t)255;
        return p;
    };
    int*    meta      = (int*)carve(1024);
    bf16_t* zeroRow   = (bf16_t*)carve((size_t)h * 2);
    bf16_t* xb        = (bf16_t*)carve((size_t)N * d * 2);
    bf16_t* w1b       = (bf16_t*)carve((size_t)E * h * d * 2);
    bf16_t* w2b       = (bf16_t*)carve((size_t)E * h * h * 2);
    bf16_t* w3b       = (bf16_t*)carve((size_t)E * d * h * 2);
    float*  gatesPair = (float*)carve((size_t)N2 * 4);
    int*    expertPr  = (int*)carve((size_t)N2 * 4);
    int*    tokSlots  = (int*)carve((size_t)N2 * 4);
    int*    tokList   = (int*)carve((size_t)padCap * 4);
    bf16_t* H1        = (bf16_t*)carve((size_t)padCap * h * 2);
    bf16_t* H2        = (bf16_t*)carve((size_t)padCap * h * 2);
    float*  Yb        = (float*)carve((size_t)padCap * d * 4);
    float*  usage     = (float*)(meta + 64);
    (void)ws_size; (void)n_in;

    // 1) init
    moe_init_meta<<<1, 128, 0, stream>>>(meta);
    moe_fill<<<(padCap + 255) / 256, 256, 0, stream>>>(tokList, zeroRow, padCap);

    // 2) fp32 -> bf16 conversions
    moe_f32_to_bf16<<<2048, 256, 0, stream>>>(x,  xb,  (size_t)N * d);
    moe_f32_to_bf16<<<2048, 256, 0, stream>>>(w1, w1b, (size_t)E * h * d);
    moe_f32_to_bf16<<<4096, 256, 0, stream>>>(w2, w2b, (size_t)E * h * h);
    moe_f32_to_bf16<<<2048, 256, 0, stream>>>(w3, w3b, (size_t)E * d * h);

    // 3) router (wave32 per token), counts via int atomics
    moe_router<<<(N + 7) / 8, 256, 0, stream>>>(x, gw, gatesPair, expertPr, meta, N);

    // 4) scan + compaction
    moe_scan<<<1, 32, 0, stream>>>(meta);
    moe_assign<<<(N2 + 255) / 256, 256, 0, stream>>>(expertPr, meta, tokList, tokSlots, N2);

    // 5) expert FFN: grouped GEMMs, 64-row blocks; maxBlocks covers worst-case padding
    const int maxBlocks = N2 / ROWS_BLK + E;
    // H1 = gelu(x @ w1^T + b1) : K=256, N=1024, gathered x rows, 4Mx4N per wave
    moe_gemm<D_DIM, H_DIM, 4, true,  true,  true ><<<maxBlocks, 256, 0, stream>>>(
        xb, w1b, b1, (void*)H1, tokList, meta, zeroRow);
    // H2 = gelu(H1 @ w2^T + b2) : K=1024, N=1024, contiguous rows, 4Mx4N per wave
    moe_gemm<H_DIM, H_DIM, 4, true,  false, true ><<<maxBlocks, 256, 0, stream>>>(
        H1, w2b, b2, (void*)H2, tokList, meta, zeroRow);
    // Y = H2 @ w3^T + b3 : K=1024, N=256, fp32 out, 4Mx2N per wave
    moe_gemm<H_DIM, D_DIM, 2, false, false, false><<<maxBlocks, 256, 0, stream>>>(
        H2, w3b, b3, (void*)Yb, tokList, meta, zeroRow);

    // 6) load-balance loss (deterministic fixed-order reduction)
    moe_usage<<<E, 256, 0, stream>>>(gatesPair, expertPr, usage, N2, N);
    moe_loss<<<1, 32, 0, stream>>>(usage, ((float*)d_out) + (size_t)N * d);

    // 7) gated combine into d_out
    const int total = N * d;
    moe_combine<<<(total + 255) / 256, 256, 0, stream>>>(Yb, gatesPair, tokSlots,
                                                         (float*)d_out, total);
    (void)out_size;
}
